// Qwen3MoeSparseMoeBlock_75222057222285
// MI455X (gfx1250) — matrix operand order repaired and checked by reading, not compiled, not run on hardware
//
#include <hip/hip_runtime.h>

typedef __attribute__((ext_vector_type(16))) __bf16 v16bf;
typedef __attribute__((ext_vector_type(8)))  __bf16 v8bf;
typedef __attribute__((ext_vector_type(8)))  float  v8f;
typedef __attribute__((ext_vector_type(4)))  float  v4f;
typedef __attribute__((__vector_size__(4 * sizeof(int)))) int v4i;

#define T_TOK 512
#define H_DIM 2048
#define I_DIM 768
#define E_NUM 64
#define TOPK  8
#define CAP   256
#define LDB   40   // LDS row stride in bf16 halves (80B): 16B-aligned, staggers banks

#if __has_builtin(__builtin_amdgcn_global_load_async_to_lds_b128) && \
    __has_builtin(__builtin_amdgcn_s_wait_asynccnt)
#define HAVE_ASYNC_LDS 1
#define GASYNC_B128(gp, lp)                                                  \
  __builtin_amdgcn_global_load_async_to_lds_b128(                            \
      (__attribute__((address_space(1))) v4i*)(gp),                          \
      (__attribute__((address_space(3))) v4i*)(lp), 0, 0)
#else
#define HAVE_ASYNC_LDS 0
#endif

__device__ __forceinline__ v16bf cat8(v8bf lo, v8bf hi) {
  return __builtin_shufflevector(lo, hi, 0,1,2,3,4,5,6,7,8,9,10,11,12,13,14,15);
}

// ---------------- kernel 1: zero output region + expert counters ----------------
__global__ void zero_out_kernel(float* __restrict__ out, int* __restrict__ counts) {
  size_t gid = (size_t)blockIdx.x * blockDim.x + threadIdx.x;   // 1024*256 = 262144 float4s
  v4f z = {0.f, 0.f, 0.f, 0.f};
  ((v4f*)out)[gid] = z;
  if (blockIdx.x == 0 && threadIdx.x < E_NUM) counts[threadIdx.x] = 0;
}

// ---------------- kernel 2: router logits (fp32, written into d_out tail) ----------------
__global__ __launch_bounds__(256) void router_logits_kernel(
    const float* __restrict__ x, const float* __restrict__ gw, float* __restrict__ logits) {
  int t = blockIdx.x;
  __shared__ float xs[H_DIM];
  for (int i = threadIdx.x; i < H_DIM; i += 256) xs[i] = x[(size_t)t * H_DIM + i];
  __syncthreads();
  int lane = threadIdx.x & 31, wave = threadIdx.x >> 5;
  for (int i = 0; i < 8; i++) {
    int e = wave * 8 + i;
    const float* g = gw + (size_t)e * H_DIM;
    float s = 0.f;
    for (int k = lane; k < H_DIM; k += 32) s += xs[k] * g[k];
    for (int off = 16; off > 0; off >>= 1) s += __shfl_xor(s, off, 32);
    if (lane == 0) logits[(size_t)t * E_NUM + e] = s;
  }
}

// ---------------- kernel 3: softmax + top-8 + renorm + dispatch ----------------
__global__ void topk_dispatch_kernel(const float* __restrict__ logits,
                                     int* __restrict__ counts,
                                     int* __restrict__ rows,
                                     float* __restrict__ roww) {
  int t = blockIdx.x;
  __shared__ float p[E_NUM];
  if (threadIdx.x < E_NUM) p[threadIdx.x] = logits[(size_t)t * E_NUM + threadIdx.x];
  __syncthreads();
  if (threadIdx.x == 0) {
    float mx = -3.4e38f;
    for (int e = 0; e < E_NUM; e++) mx = fmaxf(mx, p[e]);
    float sum = 0.f;
    for (int e = 0; e < E_NUM; e++) { float pe = __expf(p[e] - mx); p[e] = pe; sum += pe; }
    float inv = 1.f / sum;
    int   selk[TOPK]; float rwk[TOPK]; float wsum = 0.f;
    for (int k = 0; k < TOPK; k++) {
      int best = 0; float bv = -1.f;
      for (int e = 0; e < E_NUM; e++) { float v = p[e]; if (v > bv) { bv = v; best = e; } }
      selk[k] = best; rwk[k] = bv * inv; wsum += bv * inv; p[best] = -2.f;
    }
    float invw = 1.f / wsum;
    for (int k = 0; k < TOPK; k++) {
      int e = selk[k];
      int slot = atomicAdd(&counts[e], 1);
      if (slot < CAP) { rows[e * CAP + slot] = t; roww[e * CAP + slot] = rwk[k] * invw; }
    }
  }
}

// ---------------- kernel 4: gate/up GEMM + SwiGLU -> bf16 act ----------------
// Tile: M=128 (covers any realistic expert count in one pass -> weights read once),
// N=64, K-step 32. 8 waves, each owns a 32x32 sub-tile (2x2 fragments, 8 WMMA/step).
__global__ __launch_bounds__(256) void gateup_swiglu_kernel(
    const float* __restrict__ x, const float* __restrict__ w1, const float* __restrict__ w3,
    const int* __restrict__ counts, const int* __restrict__ rows, __bf16* __restrict__ act) {
  const int e = blockIdx.z;
  const int mbase = blockIdx.y * 128;
  const int n0 = blockIdx.x * 64;
  const int cnt = min(counts[e], CAP);
  if (mbase >= cnt) return;

  __shared__ __align__(16) __bf16 As[128 * LDB];
  __shared__ __align__(16) __bf16 B1[64 * LDB];
  __shared__ __align__(16) __bf16 B3[64 * LDB];
  __shared__ int rowsS[128];

  const int tid = threadIdx.x;
  if (tid < 128) { int r = mbase + tid; rowsS[tid] = (r < cnt) ? rows[e * CAP + r] : -1; }
  __syncthreads();

  const int lane = tid & 31, wave = tid >> 5;
  const int wm = wave & 3, wn = wave >> 2;
  const int hi = lane >> 4, nl = lane & 15;

  // hoisted per-thread A-row pointers; invalid rows read token 0 (always valid memory)
  // and get zeroed branchlessly after the load (no exec-mask diamonds in the hot loop).
  const float* aptr[4];
  bool avalid[4];
#pragma unroll
  for (int i = 0; i < 4; i++) {
    int r = (tid >> 3) + 32 * i;
    int tok = rowsS[r];
    avalid[i] = tok >= 0;
    aptr[i] = x + (size_t)(avalid[i] ? tok : 0) * H_DIM + (tid & 7) * 4;
  }
  const float* w1e = w1 + ((size_t)e * I_DIM + n0) * H_DIM;
  const float* w3e = w3 + ((size_t)e * I_DIM + n0) * H_DIM;

  v8f cg[2][2] = {}, cu[2][2] = {};

  for (int k0 = 0; k0 < H_DIM; k0 += 32) {
    __syncthreads();
#pragma unroll
    for (int i = 0; i < 4; i++) {          // A tile 128x32 fp32 -> bf16 (branchless zero-fill)
      int r = (tid >> 3) + 32 * i, c = (tid & 7) * 4;
      v4f av = *(const v4f*)(aptr[i] + k0);
      float m = avalid[i] ? 1.f : 0.f;
      __bf16* ad = &As[r * LDB + c];
      ad[0] = (__bf16)(av[0] * m); ad[1] = (__bf16)(av[1] * m);
      ad[2] = (__bf16)(av[2] * m); ad[3] = (__bf16)(av[3] * m);
    }
#pragma unroll
    for (int i = 0; i < 2; i++) {          // B1/B3 tiles 64x32 fp32 -> bf16
      int lin = tid + i * 256;
      int r = lin >> 3, c = (lin & 7) * 4;
      const float* p1 = w1e + (size_t)r * H_DIM + k0 + c;
      const float* p3 = w3e + (size_t)r * H_DIM + k0 + c;
      v4f b1v = *(const v4f*)p1;
      v4f b3v = *(const v4f*)p3;
      if (k0 + 32 < H_DIM) { __builtin_prefetch(p1 + 32, 0, 3); __builtin_prefetch(p3 + 32, 0, 3); }
      __bf16* b1d = &B1[r * LDB + c];
      b1d[0] = (__bf16)b1v[0]; b1d[1] = (__bf16)b1v[1]; b1d[2] = (__bf16)b1v[2]; b1d[3] = (__bf16)b1v[3];
      __bf16* b3d = &B3[r * LDB + c];
      b3d[0] = (__bf16)b3v[0]; b3d[1] = (__bf16)b3v[1]; b3d[2] = (__bf16)b3v[2]; b3d[3] = (__bf16)b3v[3];
    }
    __syncthreads();
    const int kb = hi ? 8 : 0;
    v16bf a[2];
#pragma unroll
    for (int f = 0; f < 2; f++) {
      int am = wm * 32 + f * 16 + nl;
      a[f] = cat8(*(const v8bf*)&As[am * LDB + (hi ? 8 : 0)],
                  *(const v8bf*)&As[am * LDB + (hi ? 24 : 16)]);
    }
#pragma unroll
    for (int s = 0; s < 2; s++) {
      int bn = wn * 32 + s * 16 + nl;
      v16bf b1f = cat8(*(const v8bf*)&B1[bn * LDB + kb], *(const v8bf*)&B1[bn * LDB + kb + 16]);
      v16bf b3f = cat8(*(const v8bf*)&B3[bn * LDB + kb], *(const v8bf*)&B3[bn * LDB + kb + 16]);
#pragma unroll
      for (int f = 0; f < 2; f++) {
        cg[f][s] = __builtin_amdgcn_wmma_f32_16x16x32_bf16(false, a[f], false, b1f, (short)0, cg[f][s], false, false);
        cu[f][s] = __builtin_amdgcn_wmma_f32_16x16x32_bf16(false, a[f], false, b3f, (short)0, cu[f][s], false, false);
      }
    }
  }
  // epilogue: silu(g)*u -> bf16 act[e][row][col] (invalid rows compute exact 0)
#pragma unroll
  for (int f = 0; f < 2; f++) {
#pragma unroll
    for (int v = 0; v < 8; v++) {
      int rIdx = mbase + wm * 32 + f * 16 + v + hi * 8;  // C layout: VGPR v -> M = v + 8*(lane>=16)
      __bf16* arow = act + ((size_t)e * CAP + rIdx) * I_DIM + n0 + wn * 32 + nl;
#pragma unroll
      for (int s = 0; s < 2; s++) {
        float g = cg[f][s][v], u = cu[f][s][v];
        float val = (g / (1.f + __expf(-g))) * u;
        arow[s * 16] = (__bf16)val;
      }
    }
  }
}

// ---------------- kernel 5: down GEMM + weighted scatter-combine ----------------
__global__ __launch_bounds__(256) void down_combine_kernel(
    const __bf16* __restrict__ act, const float* __restrict__ w2,
    const int* __restrict__ counts, const int* __restrict__ rows,
    const float* __restrict__ roww, float* __restrict__ out) {
  const int e = blockIdx.z;
  const int mbase = blockIdx.y * 128;
  const int n0 = blockIdx.x * 64;
  const int cnt = min(counts[e], CAP);
  if (mbase >= cnt) return;

  __shared__ __align__(16) __bf16 As[128 * LDB];
  __shared__ __align__(16) __bf16 Bs[64 * LDB];
  __shared__ int   rowsS[128];
  __shared__ float wS[128];

  const int tid = threadIdx.x;
  if (tid < 128) {
    int r = mbase + tid; bool v = r < cnt;
    rowsS[tid] = v ? rows[e * CAP + r] : -1;
    wS[tid]    = v ? roww[e * CAP + r] : 0.f;
  }
  __syncthreads();

  const int lane = tid & 31, wave = tid >> 5;
  const int wm = wave & 3, wn = wave >> 2;
  const int hi = lane >> 4, nl = lane & 15;

  const __bf16* acte = act + ((size_t)e * CAP + mbase) * I_DIM;
  const float*  w2e  = w2 + ((size_t)e * H_DIM + n0) * I_DIM;

  v8f cc[2][2] = {};

  for (int k0 = 0; k0 < I_DIM; k0 += 32) {
    __syncthreads();
#if HAVE_ASYNC_LDS
#pragma unroll
    for (int i = 0; i < 2; i++) {   // A tile 128x32 bf16: async copy global -> LDS (ASYNCcnt)
      int lin = tid + i * 256;
      int r = lin >> 2, c = (lin & 3) * 8;
      GASYNC_B128(acte + (size_t)r * I_DIM + k0 + c, &As[r * LDB + c]);
    }
#else
#pragma unroll
    for (int i = 0; i < 2; i++) {   // A tile 128x32 bf16: sync copy
      int lin = tid + i * 256;
      int r = lin >> 2, c = (lin & 3) * 8;
      *(v8bf*)&As[r * LDB + c] = *(const v8bf*)(acte + (size_t)r * I_DIM + k0 + c);
    }
#endif
#pragma unroll
    for (int i = 0; i < 2; i++) {   // B tile 64x32 fp32 -> bf16
      int lin = tid + i * 256;
      int r = lin >> 3, c = (lin & 7) * 4;
      const float* pb = w2e + (size_t)r * I_DIM + k0 + c;
      v4f bv = *(const v4f*)pb;
      if (k0 + 32 < I_DIM) __builtin_prefetch(pb + 32, 0, 3);
      __bf16* bd = &Bs[r * LDB + c];
      bd[0] = (__bf16)bv[0]; bd[1] = (__bf16)bv[1]; bd[2] = (__bf16)bv[2]; bd[3] = (__bf16)bv[3];
    }
#if HAVE_ASYNC_LDS
    __builtin_amdgcn_s_wait_asynccnt(0);
#endif
    __syncthreads();
    const int kb = hi ? 8 : 0;
    v16bf a[2];
#pragma unroll
    for (int f = 0; f < 2; f++) {
      int am = wm * 32 + f * 16 + nl;
      a[f] = cat8(*(const v8bf*)&As[am * LDB + (hi ? 8 : 0)],
                  *(const v8bf*)&As[am * LDB + (hi ? 24 : 16)]);
    }
#pragma unroll
    for (int s = 0; s < 2; s++) {
      int bn = wn * 32 + s * 16 + nl;
      v16bf b = cat8(*(const v8bf*)&Bs[bn * LDB + kb], *(const v8bf*)&Bs[bn * LDB + kb + 16]);
#pragma unroll
      for (int f = 0; f < 2; f++)
        cc[f][s] = __builtin_amdgcn_wmma_f32_16x16x32_bf16(false, a[f], false, b, (short)0, cc[f][s], false, false);
    }
  }
  // epilogue: weighted atomic scatter into out[token][col]
#pragma unroll
  for (int f = 0; f < 2; f++) {
#pragma unroll
    for (int v = 0; v < 8; v++) {
      int rl = wm * 32 + f * 16 + v + hi * 8;
      if (mbase + rl < cnt) {
        int tok = rowsS[rl];
        float wgt = wS[rl];
        float* orow = out + (size_t)tok * H_DIM + n0 + wn * 32 + nl;
#pragma unroll
        for (int s = 0; s < 2; s++)
          unsafeAtomicAdd(orow + s * 16, wgt * cc[f][s][v]);
      }
    }
  }
}

// ---------------- launch ----------------
extern "C" void kernel_launch(void* const* d_in, const int* in_sizes, int n_in,
                              void* d_out, int out_size, void* d_ws, size_t ws_size,
                              hipStream_t stream) {
  const float* x  = (const float*)d_in[0];   // [1,512,2048]
  const float* gw = (const float*)d_in[1];   // [64,2048]
  const float* w1 = (const float*)d_in[2];   // [64,768,2048]
  const float* w2 = (const float*)d_in[3];   // [64,2048,768]
  const float* w3 = (const float*)d_in[4];   // [64,768,2048]

  float* out    = (float*)d_out;                       // [512,2048]
  float* logits = out + (size_t)T_TOK * H_DIM;         // [512,64], fp32

  char* ws = (char*)d_ws;
  int*    counts = (int*)ws;                           // 64 ints
  int*    rows   = (int*)(ws + 256);                   // 64*256 ints
  float*  roww   = (float*)(ws + 256 + 65536);         // 64*256 floats
  __bf16* act    = (__bf16*)(ws + 131584);             // 64*256*768 bf16 (~25 MB)

  zero_out_kernel<<<1024, 256, 0, stream>>>(out, counts);
  router_logits_kernel<<<T_TOK, 256, 0, stream>>>(x, gw, logits);
  topk_dispatch_kernel<<<T_TOK, 64, 0, stream>>>(logits, counts, rows, roww);
  gateup_swiglu_kernel<<<dim3(I_DIM / 64, CAP / 128, E_NUM), 256, 0, stream>>>(
      x, w1, w3, counts, rows, act);
  down_combine_kernel<<<dim3(H_DIM / 64, CAP / 128, E_NUM), 256, 0, stream>>>(
      act, w2, counts, rows, roww, out);
}
